// MTMD_Skip_Lora_6227702579382
// MI455X (gfx1250) — matrix operand order repaired and checked by reading, not compiled, not run on hardware
//
#include <hip/hip_runtime.h>
#include <hip/hip_bf16.h>

typedef __bf16 bf16;
typedef bf16  v16bf __attribute__((ext_vector_type(16)));
typedef bf16  v8bf  __attribute__((ext_vector_type(8)));
typedef float v8f   __attribute__((ext_vector_type(8)));

#define B_N   32768
#define IN_N  1024
#define S1_N  1024
#define S2_N  512
#define H1_N  256
#define H2_N  64
#define D_N   3
#define T_N   2
#define E_N   8
#define R_N   16
#define DT_N  6
#define GATE_NP 64   /* gate GEMM N padded from 48 to 64 */
#define EPS_F 1e-5f
#define SKIPW 0.1f

// ---------------------------------------------------------------------------
// float -> bf16 elementwise cast
// ---------------------------------------------------------------------------
__global__ void cast_f32_bf16(const float* __restrict__ in, bf16* __restrict__ out, int n) {
    int i = blockIdx.x * 256 + threadIdx.x;
    if (i < n) out[i] = (bf16)in[i];
}

// ---------------------------------------------------------------------------
// Pack a KxN fp32 matrix into WMMA bf16 B-tile layout.
// Tiles ordered [kt][nt]; within a tile: element (lane*16 + t) holds
// B[kt*32 + (t>>3)*16 + (lane>>4)*8 + (t&7)][nt*16 + (lane&15)]  (t = vgpr*2 + pair)
// mode 0: W row-major [K][N]
// mode 1: elementwise product W * W2 (both [K][N])   (STAR effective weight)
// mode 2: grouped layout W[g][K][gs], column n = g*gs + e; zero for g >= ng
// ---------------------------------------------------------------------------
__global__ void pack_b_kernel(const float* __restrict__ W, const float* __restrict__ W2,
                              bf16* __restrict__ out, int K, int N, int mode, int gs, int ng) {
    int idx = blockIdx.x * 256 + threadIdx.x;
    int total = K * N;
    if (idx >= total) return;
    int tile = idx >> 9;
    int r    = idx & 511;
    int lane = r >> 4;
    int t    = r & 15;
    int ntiles = N >> 4;
    int kt = tile / ntiles, nt = tile - kt * ntiles;
    int k = kt * 32 + ((t >> 3) << 4) + ((lane >> 4) << 3) + (t & 7);
    int n = (nt << 4) + (lane & 15);
    float val;
    if (mode == 0)      val = W[(size_t)k * N + n];
    else if (mode == 1) val = W[(size_t)k * N + n] * W2[(size_t)k * N + n];
    else {              int g = n / gs, e = n - g * gs;
                        val = (g < ng) ? W[((size_t)g * K + k) * gs + e] : 0.f; }
    out[idx] = (bf16)val;
}

// ---------------------------------------------------------------------------
// LoRA matmul precompute: lora[dt][k][h] = sum_r a[dt][k][r] * b[dt][r][h]
// ---------------------------------------------------------------------------
__global__ void lora_mm_kernel(const float* __restrict__ a, const float* __restrict__ b,
                               float* __restrict__ out) {
    int idx = blockIdx.x * 256 + threadIdx.x;
    if (idx >= DT_N * S2_N * H2_N) return;
    int h  = idx & 63;
    int k  = (idx >> 6) & (S2_N - 1);
    int dt = idx >> 15;
    float s = 0.f;
#pragma unroll
    for (int r = 0; r < R_N; ++r)
        s += a[((size_t)dt * S2_N + k) * R_N + r] * b[((size_t)dt * R_N + r) * H2_N + h];
    out[idx] = s;
}

// ---------------------------------------------------------------------------
// Generic WMMA GEMM: C[M,N] = A[M,K] (bf16 row-major) x Bp (packed bf16 tiles)
// block = 128 threads = 4 waves. Wave w owns m-tile (blockIdx.x*4 + w) and
// accumulates 4 n-tiles, reusing one A fragment per k-step (4 WMMAs / A load;
// no duplicated A loads within a block; B tiles shared across waves via L1).
// REQUIRES: M % 64 == 0, N % 64 == 0 (callers pad N), K % 32 == 0.
// Branch-free inner loop: 2 A loads + 4 B loads + 4 WMMAs per k-step.
// mode 0: Cf[row*N+col] = acc + bias[col] (+bias2[col])     (fp32 out)
// mode 1: Cb[row*N+col] = bf16(acc + bias + bias2) only where dom[row]==dsel
// ---------------------------------------------------------------------------
__global__ __launch_bounds__(128) void gemm_bf16_wmma(
    const bf16* __restrict__ A, const bf16* __restrict__ Bp,
    int M, int K, int N,
    const float* __restrict__ bias, const float* __restrict__ bias2,
    float* __restrict__ Cf, bf16* __restrict__ Cb,
    const int* __restrict__ dom, int dsel, int mode)
{
    const int wave = threadIdx.x >> 5;
    const int lane = threadIdx.x & 31;
    const int ntiles = N >> 4;
    const int tm  = blockIdx.x * 4 + wave;          // per-wave m-tile
    const int tnb = blockIdx.y * 4;                 // base n-tile (shared by waves)
    const int m  = lane & 15;
    const int h  = lane >> 4;

    const bf16* arow = A + (size_t)(tm * 16 + m) * K + h * 8;
    const bf16* bp   = Bp + ((size_t)tnb << 9) + ((size_t)lane << 4);
    const size_t bstride = (size_t)ntiles << 9;

    v8f acc0 = {0.f,0.f,0.f,0.f,0.f,0.f,0.f,0.f};
    v8f acc1 = acc0, acc2 = acc0, acc3 = acc0;

    const int ksteps = K >> 5;
    for (int kt = 0; kt < ksteps; ++kt) {
        v8bf alo = *(const v8bf*)(arow + kt * 32);
        v8bf ahi = *(const v8bf*)(arow + kt * 32 + 16);
        v16bf av;
#pragma unroll
        for (int i = 0; i < 8; ++i) { av[i] = alo[i]; av[i + 8] = ahi[i]; }
        const bf16* bk = bp + (size_t)kt * bstride;
        v16bf b0 = *(const v16bf*)(bk);
        v16bf b1 = *(const v16bf*)(bk + 512);
        v16bf b2 = *(const v16bf*)(bk + 1024);
        v16bf b3 = *(const v16bf*)(bk + 1536);
        acc0 = __builtin_amdgcn_wmma_f32_16x16x32_bf16(false, av, false, b0, (short)0, acc0, false, false);
        acc1 = __builtin_amdgcn_wmma_f32_16x16x32_bf16(false, av, false, b1, (short)0, acc1, false, false);
        acc2 = __builtin_amdgcn_wmma_f32_16x16x32_bf16(false, av, false, b2, (short)0, acc2, false, false);
        acc3 = __builtin_amdgcn_wmma_f32_16x16x32_bf16(false, av, false, b3, (short)0, acc3, false, false);
    }

    const int row0 = tm * 16 + (h << 3);
    v8f accs[4] = {acc0, acc1, acc2, acc3};
#pragma unroll
    for (int i = 0; i < 4; ++i) {
        const int col = ((tnb + i) << 4) + m;
        float bv0 = bias ? bias[col] : 0.f;
        if (bias2) bv0 += bias2[col];
        if (mode == 0) {
#pragma unroll
            for (int r = 0; r < 8; ++r)
                Cf[(size_t)(row0 + r) * N + col] = accs[i][r] + bv0;
        } else {
#pragma unroll
            for (int r = 0; r < 8; ++r) {
                int row = row0 + r;
                if (dom[row] == dsel)
                    Cb[(size_t)row * N + col] = (bf16)(accs[i][r] + bv0);
            }
        }
    }
}

// ---------------------------------------------------------------------------
// emb = relu(LN(emb_pre; star_g, star_bb)) + relu(LN(skip_pre; skip_g, skip_bb))
// one block (256 threads) per row of 512; output bf16
// ---------------------------------------------------------------------------
__global__ __launch_bounds__(256) void ln_combine_kernel(
    const float* __restrict__ skip_pre, const float* __restrict__ emb_pre,
    const float* __restrict__ sg, const float* __restrict__ sbb,
    const float* __restrict__ eg, const float* __restrict__ ebb,
    bf16* __restrict__ emb_out)
{
    const int row = blockIdx.x;
    const int tid = threadIdx.x;
    const int wave = tid >> 5, lane = tid & 31;
    __shared__ float red[8][4];
    __shared__ float bc[4];
    const float* sp = skip_pre + (size_t)row * S2_N;
    const float* ep = emb_pre  + (size_t)row * S2_N;
    float s0 = sp[tid], s1 = sp[tid + 256];
    float e0 = ep[tid], e1 = ep[tid + 256];
    float v0 = s0 + s1, v1 = s0 * s0 + s1 * s1;
    float v2 = e0 + e1, v3 = e0 * e0 + e1 * e1;
    for (int off = 16; off; off >>= 1) {
        v0 += __shfl_xor(v0, off); v1 += __shfl_xor(v1, off);
        v2 += __shfl_xor(v2, off); v3 += __shfl_xor(v3, off);
    }
    if (lane == 0) { red[wave][0] = v0; red[wave][1] = v1; red[wave][2] = v2; red[wave][3] = v3; }
    __syncthreads();
    if (tid == 0) {
        float a = 0, b = 0, c = 0, d = 0;
        for (int w = 0; w < 8; ++w) { a += red[w][0]; b += red[w][1]; c += red[w][2]; d += red[w][3]; }
        bc[0] = a; bc[1] = b; bc[2] = c; bc[3] = d;
    }
    __syncthreads();
    const float inv = 1.0f / (float)S2_N;
    float ms = bc[0] * inv, vs = bc[1] * inv - ms * ms;
    float me = bc[2] * inv, ve = bc[3] * inv - me * me;
    float rs = rsqrtf(vs + EPS_F), re = rsqrtf(ve + EPS_F);
    bf16* op = emb_out + (size_t)row * S2_N;
    int c0 = tid, c1 = tid + 256;
    float o0 = fmaxf((s0 - ms) * rs * sg[c0] + sbb[c0], 0.f)
             + fmaxf((e0 - me) * re * eg[c0] + ebb[c0], 0.f);
    float o1 = fmaxf((s1 - ms) * rs * sg[c1] + sbb[c1], 0.f)
             + fmaxf((e1 - me) * re * eg[c1] + ebb[c1], 0.f);
    op[c0] = (bf16)o0;
    op[c1] = (bf16)o1;
}

// ---------------------------------------------------------------------------
// softmax over groups of 8 with gate bias; logits padded to stride 64 per row
// (cols 48+ unused). one thread per (b,dt): gates = softmax(logits + gate_b)
// ---------------------------------------------------------------------------
__global__ void gates_softmax_kernel(const float* __restrict__ logits,
                                     const float* __restrict__ gate_b,
                                     float* __restrict__ gates, int n) {
    int idx = blockIdx.x * 256 + threadIdx.x;
    if (idx >= n) return;
    int b = idx / DT_N, dt = idx - b * DT_N;
    const float* p  = logits + (size_t)b * GATE_NP + dt * E_N;
    const float* gb = gate_b + dt * E_N;
    float v[8];
#pragma unroll
    for (int e = 0; e < 8; ++e) v[e] = p[e] + gb[e];
    float mx = v[0];
#pragma unroll
    for (int e = 1; e < 8; ++e) mx = fmaxf(mx, v[e]);
    float s = 0.f;
#pragma unroll
    for (int e = 0; e < 8; ++e) { v[e] = __expf(v[e] - mx); s += v[e]; }
    float r = 1.0f / s;
    float* o = gates + (size_t)idx * 8;
#pragma unroll
    for (int e = 0; e < 8; ++e) o[e] = v[e] * r;
}

// ---------------------------------------------------------------------------
// Fused expert MLP: per (16-row tile, expert):
//   h   = relu(LN(emb @ W1[e] + b1))   (WMMA, K=512 -> 256 cols)
//   fea = relu(LN(h   @ W2[e] + b2))   (WMMA from LDS, K=256 -> 64 cols)
// block = 256 threads = 8 waves
// ---------------------------------------------------------------------------
__global__ __launch_bounds__(256) void experts_fused_kernel(
    const bf16* __restrict__ emb, const bf16* __restrict__ W1p, const bf16* __restrict__ W2p,
    const float* __restrict__ b1, const float* __restrict__ g1, const float* __restrict__ bb1,
    const float* __restrict__ b2, const float* __restrict__ g2, const float* __restrict__ bb2,
    float* __restrict__ fea)
{
    const int e  = blockIdx.y;
    const int tm = blockIdx.x;
    const int wave = threadIdx.x >> 5, lane = threadIdx.x & 31;
    const int m = lane & 15, h = lane >> 4;
    __shared__ float hs[16 * 256];
    __shared__ bf16  hb[16 * 256];
    __shared__ float mu[16], rsd[16];

    // ---- stage 1: [16,512] @ [512,256] ; each wave does 2 n-tiles ----
    const bf16* arow = emb + (size_t)(tm * 16 + m) * S2_N + h * 8;
    const bf16* W1e  = W1p + (size_t)e * (S2_N * H1_N);
    const int tn0 = wave * 2, tn1 = tn0 + 1;
    v8f a0 = {0.f,0.f,0.f,0.f,0.f,0.f,0.f,0.f};
    v8f a1 = {0.f,0.f,0.f,0.f,0.f,0.f,0.f,0.f};
    for (int kt = 0; kt < 16; ++kt) {
        v8bf alo = *(const v8bf*)(arow + kt * 32);
        v8bf ahi = *(const v8bf*)(arow + kt * 32 + 16);
        v16bf av;
#pragma unroll
        for (int i = 0; i < 8; ++i) { av[i] = alo[i]; av[i + 8] = ahi[i]; }
        v16bf b0 = *(const v16bf*)(W1e + (((size_t)kt * 16 + tn0) << 9) + ((size_t)lane << 4));
        v16bf b1v = *(const v16bf*)(W1e + (((size_t)kt * 16 + tn1) << 9) + ((size_t)lane << 4));
        a0 = __builtin_amdgcn_wmma_f32_16x16x32_bf16(false, av, false, b0, (short)0, a0, false, false);
        a1 = __builtin_amdgcn_wmma_f32_16x16x32_bf16(false, av, false, b1v, (short)0, a1, false, false);
    }
    const int mb = h << 3;
    {
        int c0 = (tn0 << 4) + m, c1 = (tn1 << 4) + m;
#pragma unroll
        for (int r = 0; r < 8; ++r) {
            hs[(mb + r) * 256 + c0] = a0[r] + b1[e * H1_N + c0];
            hs[(mb + r) * 256 + c1] = a1[r] + b1[e * H1_N + c1];
        }
    }
    __syncthreads();
    if (threadIdx.x < 16) {
        float s = 0.f, s2 = 0.f;
        const float* rp = hs + threadIdx.x * 256;
        for (int c = 0; c < 256; ++c) { float v = rp[c]; s += v; s2 += v * v; }
        float mm = s / 256.f, vv = s2 / 256.f - mm * mm;
        mu[threadIdx.x] = mm; rsd[threadIdx.x] = rsqrtf(vv + EPS_F);
    }
    __syncthreads();
#pragma unroll
    for (int i = 0; i < 16; ++i) {
        int c = threadIdx.x;
        float v = (hs[i * 256 + c] - mu[i]) * rsd[i] * g1[e * H1_N + c] + bb1[e * H1_N + c];
        hb[i * 256 + c] = (bf16)fmaxf(v, 0.f);
    }
    __syncthreads();

    // ---- stage 2: [16,256] @ [256,64] ; waves 0..3, one n-tile each ----
    if (wave < 4) {
        const bf16* arow2 = hb + m * 256 + h * 8;
        const bf16* W2e = W2p + (size_t)e * (H1_N * H2_N);
        v8f c2 = {0.f,0.f,0.f,0.f,0.f,0.f,0.f,0.f};
        for (int kt = 0; kt < 8; ++kt) {
            v8bf alo = *(const v8bf*)(arow2 + kt * 32);
            v8bf ahi = *(const v8bf*)(arow2 + kt * 32 + 16);
            v16bf av;
#pragma unroll
            for (int i = 0; i < 8; ++i) { av[i] = alo[i]; av[i + 8] = ahi[i]; }
            v16bf bv = *(const v16bf*)(W2e + (((size_t)kt * 4 + wave) << 9) + ((size_t)lane << 4));
            c2 = __builtin_amdgcn_wmma_f32_16x16x32_bf16(false, av, false, bv, (short)0, c2, false, false);
        }
        int c = (wave << 4) + m;
#pragma unroll
        for (int r = 0; r < 8; ++r)
            hs[(mb + r) * 64 + c] = c2[r] + b2[e * H2_N + c];
    }
    __syncthreads();
    if (threadIdx.x < 16) {
        float s = 0.f, s2 = 0.f;
        const float* rp = hs + threadIdx.x * 64;
        for (int c = 0; c < 64; ++c) { float v = rp[c]; s += v; s2 += v * v; }
        float mm = s / 64.f, vv = s2 / 64.f - mm * mm;
        mu[threadIdx.x] = mm; rsd[threadIdx.x] = rsqrtf(vv + EPS_F);
    }
    __syncthreads();
#pragma unroll
    for (int i = 0; i < 4; ++i) {
        int j = i * 256 + threadIdx.x;
        int row = j >> 6, c = j & 63;
        float v = (hs[j] - mu[row]) * rsd[row] * g2[e * H2_N + c] + bb2[e * H2_N + c];
        fea[((size_t)(tm * 16 + row) * E_N + e) * H2_N + c] = fmaxf(v, 0.f);
    }
}

// ---------------------------------------------------------------------------
// Towers: per sample b, only the two dt = t*D + domain[b] heads.
// ---------------------------------------------------------------------------
__global__ __launch_bounds__(64) void towers_kernel(
    const float* __restrict__ gates, const float* __restrict__ fea,
    const float* __restrict__ lora_out, const int* __restrict__ dom,
    const float* __restrict__ tw_W1, const float* __restrict__ tw_b1,
    const float* __restrict__ tw_g, const float* __restrict__ tw_bb,
    const float* __restrict__ tw_W2, const float* __restrict__ tw_b2,
    float* __restrict__ out)
{
    const int b = blockIdx.x;
    const int j = threadIdx.x;
    __shared__ float sh[64], sh2[64], bc[2];
    const int d = dom[b];
    for (int t = 0; t < T_N; ++t) {
        const int dt = t * D_N + d;
        float tf = SKIPW * lora_out[((size_t)b * DT_N + dt) * H2_N + j];
        const float* gp = gates + (size_t)b * (DT_N * E_N) + dt * E_N;
#pragma unroll
        for (int e = 0; e < E_N; ++e)
            tf += gp[e] * fea[((size_t)b * E_N + e) * H2_N + j];
        sh[j] = tf;
        __syncthreads();
        float y = tw_b1[dt * H2_N + j];
        for (int k = 0; k < H2_N; ++k)
            y += sh[k] * tw_W1[((size_t)dt * H2_N + k) * H2_N + j];
        sh2[j] = y;
        __syncthreads();
        if (j == 0) {
            float s = 0.f, s2 = 0.f;
            for (int k = 0; k < 64; ++k) { float v = sh2[k]; s += v; s2 += v * v; }
            float mm = s / 64.f, vv = s2 / 64.f - mm * mm;
            bc[0] = mm; bc[1] = rsqrtf(vv + EPS_F);
        }
        __syncthreads();
        float th = fmaxf((y - bc[0]) * bc[1] * tw_g[dt * H2_N + j] + tw_bb[dt * H2_N + j], 0.f);
        sh[j] = th * tw_W2[dt * H2_N + j];
        __syncthreads();
        if (j == 0) {
            float s = 0.f;
            for (int k = 0; k < 64; ++k) s += sh[k];
            s += tw_b2[dt];
            out[(size_t)b * T_N + t] = 1.0f / (1.0f + __expf(-s));
        }
        __syncthreads();
    }
}

// ---------------------------------------------------------------------------
extern "C" void kernel_launch(void* const* d_in, const int* in_sizes, int n_in,
                              void* d_out, int out_size, void* d_ws, size_t ws_size,
                              hipStream_t stream) {
    (void)in_sizes; (void)n_in; (void)out_size; (void)ws_size;
    const float* input_emb = (const float*)d_in[0];
    const int*   domain_id = (const int*)d_in[1];
    const float* skip_W  = (const float*)d_in[2];
    const float* skip_b  = (const float*)d_in[3];
    const float* skip_g  = (const float*)d_in[4];
    const float* skip_bb = (const float*)d_in[5];
    const float* shared_W = (const float*)d_in[6];
    const float* shared_b = (const float*)d_in[7];
    const float* slot_W   = (const float*)d_in[8];
    const float* slot_b   = (const float*)d_in[9];
    const float* star_W  = (const float*)d_in[10];
    const float* star_b  = (const float*)d_in[11];
    const float* star_g  = (const float*)d_in[12];
    const float* star_bb = (const float*)d_in[13];
    const float* e_W1  = (const float*)d_in[14];
    const float* e_b1  = (const float*)d_in[15];
    const float* e_g1  = (const float*)d_in[16];
    const float* e_bb1 = (const float*)d_in[17];
    const float* e_W2  = (const float*)d_in[18];
    const float* e_b2  = (const float*)d_in[19];
    const float* e_g2  = (const float*)d_in[20];
    const float* e_bb2 = (const float*)d_in[21];
    const float* gate_W = (const float*)d_in[22];
    const float* gate_b = (const float*)d_in[23];
    const float* lora_a = (const float*)d_in[24];
    const float* lora_b = (const float*)d_in[25];
    const float* tw_W1 = (const float*)d_in[26];
    const float* tw_b1 = (const float*)d_in[27];
    const float* tw_g  = (const float*)d_in[28];
    const float* tw_bb = (const float*)d_in[29];
    const float* tw_W2 = (const float*)d_in[30];
    const float* tw_b2 = (const float*)d_in[31];

    // ---- workspace layout ----
    char* ws = (char*)d_ws;
    size_t off = 0;
    auto alloc = [&](size_t bytes) -> char* {
        char* p = ws + off;
        off = (off + bytes + 255) & ~(size_t)255;
        return p;
    };
    bf16*  inA    = (bf16*)alloc((size_t)B_N * IN_N * 2);       // dead after STAR -> reuse for loraOut
    bf16*  pSkipW = (bf16*)alloc((size_t)IN_N * S2_N * 2);
    bf16*  pEffW  = (bf16*)alloc((size_t)D_N * IN_N * S1_N * 2);
    bf16*  pStarW = (bf16*)alloc((size_t)S1_N * S2_N * 2);
    bf16*  pEW1   = (bf16*)alloc((size_t)E_N * S2_N * H1_N * 2);
    bf16*  pEW2   = (bf16*)alloc((size_t)E_N * H1_N * H2_N * 2);
    bf16*  pGateW = (bf16*)alloc((size_t)S2_N * GATE_NP * 2);   // padded 48 -> 64 cols
    float* loraF  = (float*)alloc((size_t)DT_N * S2_N * H2_N * 4);
    bf16*  pLoraW = (bf16*)alloc((size_t)S2_N * DT_N * H2_N * 2);
    float* skipPre = (float*)alloc((size_t)B_N * S2_N * 4);     // dead after ln_combine -> reuse for fea
    bf16*  starB   = (bf16*)alloc((size_t)B_N * S1_N * 2);
    float* embPre  = (float*)alloc((size_t)B_N * S2_N * 4);
    bf16*  embB    = (bf16*)alloc((size_t)B_N * S2_N * 2);
    float* logits  = (float*)alloc((size_t)B_N * GATE_NP * 4);  // padded stride 64
    float* gatesW  = (float*)alloc((size_t)B_N * DT_N * E_N * 4);
    float* fea     = skipPre;                 // alias: B*E*H2*4 == B*S2*4
    float* loraOut = (float*)inA;             // alias: B*DT*H2*4 < B*IN*2

    // ---- 1. cast input to bf16 ----
    {
        int n = B_N * IN_N;
        cast_f32_bf16<<<(n + 255) / 256, 256, 0, stream>>>(input_emb, inA, n);
    }
    // ---- 2. pack weights ----
    auto pack = [&](const float* W, const float* W2, bf16* dst, int K, int N, int mode, int gs, int ng) {
        int n = K * N;
        pack_b_kernel<<<(n + 255) / 256, 256, 0, stream>>>(W, W2, dst, K, N, mode, gs, ng);
    };
    pack(skip_W, nullptr, pSkipW, IN_N, S2_N, 0, 0, 0);
    for (int d = 0; d < D_N; ++d)
        pack(slot_W + (size_t)d * IN_N * S1_N, shared_W,
             pEffW + (size_t)d * IN_N * S1_N, IN_N, S1_N, 1, 0, 0);
    pack(star_W, nullptr, pStarW, S1_N, S2_N, 0, 0, 0);
    for (int e = 0; e < E_N; ++e) {
        pack(e_W1 + (size_t)e * S2_N * H1_N, nullptr, pEW1 + (size_t)e * S2_N * H1_N, S2_N, H1_N, 0, 0, 0);
        pack(e_W2 + (size_t)e * H1_N * H2_N, nullptr, pEW2 + (size_t)e * H1_N * H2_N, H1_N, H2_N, 0, 0, 0);
    }
    pack(gate_W, nullptr, pGateW, S2_N, GATE_NP, 2, E_N, DT_N);   // cols 48..63 zero
    // ---- 3. LoRA precompute + pack ----
    {
        int n = DT_N * S2_N * H2_N;
        lora_mm_kernel<<<(n + 255) / 256, 256, 0, stream>>>(lora_a, lora_b, loraF);
    }
    pack(loraF, nullptr, pLoraW, S2_N, DT_N * H2_N, 2, H2_N, DT_N);

    // ---- 4. skip GEMM: skipPre = input @ skip_W + skip_b ----
    gemm_bf16_wmma<<<dim3(B_N / 64, S2_N / 64), 128, 0, stream>>>(
        inA, pSkipW, B_N, IN_N, S2_N, skip_b, nullptr, skipPre, nullptr, nullptr, 0, 0);
    // ---- 5. STAR: 3 domain GEMMs with row-masked bf16 epilogue ----
    for (int d = 0; d < D_N; ++d)
        gemm_bf16_wmma<<<dim3(B_N / 64, S1_N / 64), 128, 0, stream>>>(
            inA, pEffW + (size_t)d * IN_N * S1_N, B_N, IN_N, S1_N,
            slot_b + (size_t)d * S1_N, shared_b, nullptr, starB, domain_id, d, 1);
    // ---- 6. star_mlp GEMM: embPre = star @ star_W + star_b ----
    gemm_bf16_wmma<<<dim3(B_N / 64, S2_N / 64), 128, 0, stream>>>(
        starB, pStarW, B_N, S1_N, S2_N, star_b, nullptr, embPre, nullptr, nullptr, 0, 0);
    // ---- 7. emb = relu(LN(embPre)) + relu(LN(skipPre)) -> bf16 ----
    ln_combine_kernel<<<B_N, 256, 0, stream>>>(skipPre, embPre, skip_g, skip_bb,
                                               star_g, star_bb, embB);
    // ---- 8. gate logits GEMM [B,64 padded]; bias added in softmax ----
    gemm_bf16_wmma<<<dim3(B_N / 64, GATE_NP / 64), 128, 0, stream>>>(
        embB, pGateW, B_N, S2_N, GATE_NP, nullptr, nullptr, logits, nullptr, nullptr, 0, 0);
    {
        int n = B_N * DT_N;
        gates_softmax_kernel<<<(n + 255) / 256, 256, 0, stream>>>(logits, gate_b, gatesW, n);
    }
    // ---- 9. LoRA-out GEMM: loraOut[B, DT*H2] = emb @ lora ----
    gemm_bf16_wmma<<<dim3(B_N / 64, (DT_N * H2_N) / 64), 128, 0, stream>>>(
        embB, pLoraW, B_N, S2_N, DT_N * H2_N, nullptr, nullptr, loraOut, nullptr, nullptr, 0, 0);
    // ---- 10. fused experts -> fea [B,E,H2] ----
    experts_fused_kernel<<<dim3(B_N / 16, E_N), 256, 0, stream>>>(
        embB, pEW1, pEW2, e_b1, e_g1, e_bb1, e_b2, e_g2, e_bb2, fea);
    // ---- 11. towers + domain select -> out [B,T] ----
    towers_kernel<<<B_N, 64, 0, stream>>>(gatesW, fea, loraOut, domain_id,
                                          tw_W1, tw_b1, tw_g, tw_bb, tw_W2, tw_b2,
                                          (float*)d_out);
}
